// GCAT_9431748182564
// MI455X (gfx1250) — matrix operand order repaired and checked by reading, not compiled, not run on hardware
//
#include <hip/hip_runtime.h>

// ---- problem constants ----
#define GN    8192     // nodes
#define FIN   512      // input features
#define FO    200      // F_OUT
#define FOP   208      // F_OUT padded to 13*16
#define NTIL  13       // FOP/16
#define CN    20       // classes
#define CNP   32       // padded classes (2 N-tiles)
#define ALPHA 0.2f

typedef __attribute__((ext_vector_type(2)))  float   v2f;
typedef __attribute__((ext_vector_type(8)))  float   v8f;
typedef __attribute__((ext_vector_type(16))) __bf16  v16bf;

// -------------------------------------------------------------------------
// Kernel 0: pad + transpose weights so all WMMA B-fragment loads are
// unconditional contiguous b64 loads.  WT[FOP][FIN], W0T[CNP][FOP].
// -------------------------------------------------------------------------
__global__ void gat_prep(const float* __restrict__ W, const float* __restrict__ W0,
                         float* __restrict__ WT, float* __restrict__ W0T) {
    int idx = blockIdx.x * 256 + threadIdx.x;
    if (idx < FOP * FIN) {
        int col = idx / FIN, k = idx % FIN;
        WT[idx] = (col < FO) ? W[(size_t)k * FO + col] : 0.f;
    }
    if (idx < CNP * FOP) {
        int col = idx / FOP, k = idx % FOP;
        W0T[idx] = (col < CN && k < FO) ? W0[(size_t)k * CN + col] : 0.f;
    }
}

// -------------------------------------------------------------------------
// Kernel 1: h = x @ W  (fp32 WMMA 16x16x4), also store hT in bf16 transposed
// grid (512, 13), block 32 (one wave per 16x16 tile of h)
// -------------------------------------------------------------------------
__global__ void gat_h(const float* __restrict__ x, const float* __restrict__ WT,
                      float* __restrict__ h, __bf16* __restrict__ hT) {
    int mt = blockIdx.x, nt = blockIdx.y;
    int lane = threadIdx.x;
    int half = lane >> 4, lm = lane & 15;
    int m0 = mt * 16, n0 = nt * 16;
    int col = n0 + lm;
    const float* xr = x  + (size_t)(m0 + lm) * FIN + 2 * half;   // A: rows
    const float* wc = WT + (size_t)col       * FIN + 2 * half;   // B: cols (padded)

    v8f c = {};
    for (int k = 0; k < FIN; k += 4) {
        v2f a = *(const v2f*)(xr + k);
        v2f b = *(const v2f*)(wc + k);
        c = __builtin_amdgcn_wmma_f32_16x16x4_f32(false, a, false, b,
                                                  (short)0, c, false, false);
    }
#pragma unroll
    for (int v = 0; v < 8; v++) {
        int row = m0 + half * 8 + v;
        float val = c[v];
        h[(size_t)row * FOP + col] = val;              // padded cols get 0
        hT[(size_t)col * GN + row] = (__bf16)val;      // bf16 transposed copy
    }
}

// -------------------------------------------------------------------------
// Kernel 2: s1 = h@a1, s2 = h@a2  (one wave per row)
// grid 1024, block 256 (8 waves)
// -------------------------------------------------------------------------
__global__ void gat_s1s2(const float* __restrict__ h, const float* __restrict__ a1,
                         const float* __restrict__ a2,
                         float* __restrict__ s1, float* __restrict__ s2) {
    int wave = threadIdx.x >> 5;
    int lane = threadIdx.x & 31;
    int row  = blockIdx.x * 8 + wave;
    const float* hr = h + (size_t)row * FOP;
    float acc1 = 0.f, acc2 = 0.f;
    for (int n = lane; n < FO; n += 32) {
        float v = hr[n];
        acc1 += v * a1[n];
        acc2 += v * a2[n];
    }
#pragma unroll
    for (int off = 16; off > 0; off >>= 1) {
        acc1 += __shfl_down(acc1, off, 32);
        acc2 += __shfl_down(acc2, off, 32);
    }
    if (lane == 0) { s1[row] = acc1; s2[row] = acc2; }
}

// -------------------------------------------------------------------------
// Kernel 3: per-row online softmax stats over masked leaky-relu logits
// grid 8192, block 256; single streaming pass over support (float4 loads).
// Branchy on purpose: ~99% of entries are masked, so whole waves skip exp.
// -------------------------------------------------------------------------
__global__ void gat_rowstats(const float* __restrict__ support,
                             const float* __restrict__ s1, const float* __restrict__ s2,
                             float* __restrict__ mrow, float* __restrict__ lrow) {
    int row = blockIdx.x, tid = threadIdx.x;
    float s1r = s1[row];
    const float4* srow = (const float4*)(support + (size_t)row * GN);
    const float4* z4   = (const float4*)s2;
    float m = -9e15f, l = 0.f;
#define ONLINE(sup, z)                                            \
    { if ((sup) > 0.f) {                                          \
        float e = s1r + (z);                                      \
        e = e > 0.f ? e : ALPHA * e;                              \
        if (e > m) { l = l * __expf(m - e) + 1.f; m = e; }        \
        else       { l += __expf(e - m); } } }
    for (int j = tid; j < GN / 4; j += 256) {
        float4 sv = srow[j];
        float4 zv = z4[j];
        ONLINE(sv.x, zv.x); ONLINE(sv.y, zv.y);
        ONLINE(sv.z, zv.z); ONLINE(sv.w, zv.w);
    }
#undef ONLINE
    __shared__ float sm[256], sl[256];
    sm[tid] = m; sl[tid] = l;
    __syncthreads();
    for (int off = 128; off > 0; off >>= 1) {
        if (tid < off) {
            float ma = sm[tid], mb = sm[tid + off];
            float M = fmaxf(ma, mb);
            sl[tid] = sl[tid] * __expf(ma - M) + sl[tid + off] * __expf(mb - M);
            sm[tid] = M;
        }
        __syncthreads();
    }
    if (tid == 0) { mrow[row] = sm[0]; lrow[row] = 1.f / sl[0]; }
}

// -------------------------------------------------------------------------
// Kernel 4 (core): h' = elu(softmax(mask(e)) @ h) via bf16 WMMA 16x16x32.
// grid 512 (16-row tiles), block 128 (4 waves split K).  Branchless prob:
// masked logit -> -1e30 so exp underflows to exactly 0 (cndmask + exp).
// -------------------------------------------------------------------------
__device__ __forceinline__ float att_p(float sup, float s1r, float z,
                                       float mr, float invl) {
    float e = s1r + z;
    e = e > 0.f ? e : ALPHA * e;
    e = sup > 0.f ? e : -1e30f;            // masked -> exp() == 0
    return __expf(e - mr) * invl;
}

__global__ void __launch_bounds__(128)
gat_attn(const float* __restrict__ support, const __bf16* __restrict__ hT,
         const float* __restrict__ s1, const float* __restrict__ s2,
         const float* __restrict__ mrow, const float* __restrict__ lrow,
         float* __restrict__ hp) {
    int mt = blockIdx.x, tid = threadIdx.x;
    int wave = tid >> 5, lane = tid & 31;
    int half = lane >> 4, lm = lane & 15;
    int m0 = mt * 16;
    int row = m0 + lm;                     // A-fragment row for this lane
    float s1r  = s1[row];
    float mr   = mrow[row];
    float invl = lrow[row];
    const float* srow = support + (size_t)row * GN;

    v8f c[NTIL];
    v8f zero = {};
#pragma unroll
    for (int t = 0; t < NTIL; t++) c[t] = zero;

    for (int k = wave * 32; k < GN; k += 128) {
        int kb = k + half * 16;
        // A elements 0..15 -> K = kb+0..15 (contiguous, matches B)
        float4 sa = *(const float4*)(srow + kb);
        float4 sb = *(const float4*)(srow + kb + 4);
        float4 sc = *(const float4*)(srow + kb + 8);
        float4 sd = *(const float4*)(srow + kb + 12);
        float4 za = *(const float4*)(s2 + kb);
        float4 zb = *(const float4*)(s2 + kb + 4);
        float4 zc = *(const float4*)(s2 + kb + 8);
        float4 zd = *(const float4*)(s2 + kb + 12);
        v16bf a;
        a[0]  = (__bf16)att_p(sa.x, s1r, za.x, mr, invl);
        a[1]  = (__bf16)att_p(sa.y, s1r, za.y, mr, invl);
        a[2]  = (__bf16)att_p(sa.z, s1r, za.z, mr, invl);
        a[3]  = (__bf16)att_p(sa.w, s1r, za.w, mr, invl);
        a[4]  = (__bf16)att_p(sb.x, s1r, zb.x, mr, invl);
        a[5]  = (__bf16)att_p(sb.y, s1r, zb.y, mr, invl);
        a[6]  = (__bf16)att_p(sb.z, s1r, zb.z, mr, invl);
        a[7]  = (__bf16)att_p(sb.w, s1r, zb.w, mr, invl);
        a[8]  = (__bf16)att_p(sc.x, s1r, zc.x, mr, invl);
        a[9]  = (__bf16)att_p(sc.y, s1r, zc.y, mr, invl);
        a[10] = (__bf16)att_p(sc.z, s1r, zc.z, mr, invl);
        a[11] = (__bf16)att_p(sc.w, s1r, zc.w, mr, invl);
        a[12] = (__bf16)att_p(sd.x, s1r, zd.x, mr, invl);
        a[13] = (__bf16)att_p(sd.y, s1r, zd.y, mr, invl);
        a[14] = (__bf16)att_p(sd.z, s1r, zd.z, mr, invl);
        a[15] = (__bf16)att_p(sd.w, s1r, zd.w, mr, invl);
#pragma unroll
        for (int t = 0; t < NTIL; t++) {
            // B fragment: lanes 0-15 hold K = k..k+15, lanes 16-31 K = k+16..31
            // -> one contiguous 32-byte load from transposed bf16 h
            const __bf16* hcol = hT + (size_t)(t * 16 + lm) * GN + k + half * 16;
            v16bf b = *(const v16bf*)hcol;
            c[t] = __builtin_amdgcn_wmma_f32_16x16x32_bf16(false, a, false, b,
                                                           (short)0, c[t], false, false);
        }
    }

    __shared__ float red[4][16][FOP];      // 53,248 bytes
#pragma unroll
    for (int t = 0; t < NTIL; t++)
#pragma unroll
        for (int v = 0; v < 8; v++)
            red[wave][half * 8 + v][t * 16 + lm] = c[t][v];
    __syncthreads();
    for (int idx = tid; idx < 16 * FOP; idx += 128) {
        int r = idx / FOP, cc = idx % FOP;
        float s = red[0][r][cc] + red[1][r][cc] + red[2][r][cc] + red[3][r][cc];
        s = s > 0.f ? s : (__expf(s) - 1.f);   // ELU
        hp[(size_t)(m0 + r) * FOP + cc] = s;
    }
}

// -------------------------------------------------------------------------
// Kernel 5: tmpT = (hp @ W0)^T  (fp32 WMMA; K=208; output stored transposed
// [CNP][GN] so kernel 6's B loads are contiguous)
// grid 512, block 32
// -------------------------------------------------------------------------
__global__ void gat_hpw0(const float* __restrict__ hp, const float* __restrict__ W0T,
                         float* __restrict__ tmpT) {
    int mt = blockIdx.x;
    int lane = threadIdx.x;
    int half = lane >> 4, lm = lane & 15;
    int m0 = mt * 16;
    const float* hr  = hp  + (size_t)(m0 + lm) * FOP + 2 * half;
    const float* w0a = W0T + (size_t)lm        * FOP + 2 * half;
    const float* w0b = W0T + (size_t)(16 + lm) * FOP + 2 * half;
    v8f c0 = {}, c1 = {};
    for (int k = 0; k < FOP; k += 4) {
        v2f a  = *(const v2f*)(hr  + k);
        v2f b0 = *(const v2f*)(w0a + k);
        v2f b1 = *(const v2f*)(w0b + k);
        c0 = __builtin_amdgcn_wmma_f32_16x16x4_f32(false, a, false, b0, (short)0, c0, false, false);
        c1 = __builtin_amdgcn_wmma_f32_16x16x4_f32(false, a, false, b1, (short)0, c1, false, false);
    }
#pragma unroll
    for (int v = 0; v < 8; v++) {
        int r = m0 + half * 8 + v;
        tmpT[(size_t)lm        * GN + r] = c0[v];
        tmpT[(size_t)(16 + lm) * GN + r] = c1[v];
    }
}

// -------------------------------------------------------------------------
// Kernel 6: out = support @ tmp  (fp32 WMMA, 4-wave K-split, LDS reduce)
// grid 512, block 128; B loads contiguous from tmpT (L2-resident, 1 MB)
// -------------------------------------------------------------------------
__global__ void __launch_bounds__(128)
gat_out(const float* __restrict__ support, const float* __restrict__ tmpT,
        float* __restrict__ out) {
    int mt = blockIdx.x, tid = threadIdx.x;
    int wave = tid >> 5, lane = tid & 31;
    int half = lane >> 4, lm = lane & 15;
    int m0 = mt * 16;
    const float* srow = support + (size_t)(m0 + lm) * GN + 2 * half;
    const float* t0   = tmpT + (size_t)lm        * GN + 2 * half;
    const float* t1   = tmpT + (size_t)(16 + lm) * GN + 2 * half;
    v8f c0 = {}, c1 = {};
    for (int k = wave * 4; k < GN; k += 16) {
        v2f a  = *(const v2f*)(srow + k);
        v2f b0 = *(const v2f*)(t0 + k);
        v2f b1 = *(const v2f*)(t1 + k);
        c0 = __builtin_amdgcn_wmma_f32_16x16x4_f32(false, a, false, b0, (short)0, c0, false, false);
        c1 = __builtin_amdgcn_wmma_f32_16x16x4_f32(false, a, false, b1, (short)0, c1, false, false);
    }
    __shared__ float red[4][16][CNP];
#pragma unroll
    for (int v = 0; v < 8; v++) {
        red[wave][half * 8 + v][lm]      = c0[v];
        red[wave][half * 8 + v][16 + lm] = c1[v];
    }
    __syncthreads();
    for (int idx = tid; idx < 16 * CN; idx += 128) {
        int r = idx / CN, cc = idx % CN;
        float s = red[0][r][cc] + red[1][r][cc] + red[2][r][cc] + red[3][r][cc];
        out[(size_t)(m0 + r) * CN + cc] = s;
    }
}

// -------------------------------------------------------------------------
extern "C" void kernel_launch(void* const* d_in, const int* in_sizes, int n_in,
                              void* d_out, int out_size, void* d_ws, size_t ws_size,
                              hipStream_t stream) {
    const float* x       = (const float*)d_in[0];
    const float* support = (const float*)d_in[1];
    const float* W       = (const float*)d_in[2];
    const float* a1      = (const float*)d_in[3];
    const float* a2      = (const float*)d_in[4];
    const float* W0      = (const float*)d_in[5];
    float* out = (float*)d_out;

    // workspace layout (~18.8 MB)
    char* p = (char*)d_ws;
    float*  h    = (float*)p;            p += (size_t)GN * FOP * sizeof(float);
    __bf16* hT   = (__bf16*)p;           p += (size_t)FOP * GN * sizeof(__bf16);
    float*  s1   = (float*)p;            p += (size_t)GN * sizeof(float);
    float*  s2   = (float*)p;            p += (size_t)GN * sizeof(float);
    float*  mrow = (float*)p;            p += (size_t)GN * sizeof(float);
    float*  lrow = (float*)p;            p += (size_t)GN * sizeof(float);
    float*  hp   = (float*)p;            p += (size_t)GN * FOP * sizeof(float);
    float*  tmpT = (float*)p;            p += (size_t)CNP * GN * sizeof(float);
    float*  WT   = (float*)p;            p += (size_t)FOP * FIN * sizeof(float);
    float*  W0T  = (float*)p;            p += (size_t)CNP * FOP * sizeof(float);

    gat_prep    <<<(FOP * FIN + 255) / 256, 256, 0, stream>>>(W, W0, WT, W0T);
    gat_h       <<<dim3(GN / 16, NTIL), 32,  0, stream>>>(x, WT, h, hT);
    gat_s1s2    <<<GN / 8,              256, 0, stream>>>(h, a1, a2, s1, s2);
    gat_rowstats<<<GN,                  256, 0, stream>>>(support, s1, s2, mrow, lrow);
    gat_attn    <<<GN / 16,             128, 0, stream>>>(support, hT, s1, s2, mrow, lrow, hp);
    gat_hpw0    <<<GN / 16,             32,  0, stream>>>(hp, W0T, tmpT);
    gat_out     <<<GN / 16,             128, 0, stream>>>(support, tmpT, out);
}
